// DCLAKNet_83373905150439
// MI455X (gfx1250) — compile-verified, matrix operand order checked by reading, not run on hardware
//
#include <hip/hip_runtime.h>
#include <hip/hip_bf16.h>

typedef __attribute__((ext_vector_type(16))) __bf16 v16bf;
typedef __attribute__((ext_vector_type(8)))  float  v8f;

__device__ __forceinline__ float bf_hi_part(float f) {
    unsigned u = __builtin_bit_cast(unsigned, f) & 0xffff0000u;
    return __builtin_bit_cast(float, u);
}

// ---------------------------------------------------------------------------
// Weight prep: OIHW fp32 -> WMMA A-fragment order, pre-split into bf16 hi/lo.
// One thread per (mtile, chunk, lane); writes 2 contiguous v16bf (64 B).
// Zero-pads both the M tail (m >= Cout) and the K tail (k >= Ktot).
// ---------------------------------------------------------------------------
template<int CIN, int KSZ>
__global__ void __launch_bounds__(256)
pack_w_kernel(const float* __restrict__ w, __bf16* __restrict__ packed,
              int Cout, int Mtiles)
{
    constexpr int KK     = KSZ * KSZ;
    constexpr int Ktot   = CIN * KK;
    constexpr int nchunk = (Ktot + 31) / 32;
    int gid   = blockIdx.x * blockDim.x + threadIdx.x;
    int total = Mtiles * nchunk * 32;
    if (gid >= total) return;
    int lane  = gid & 31;
    int rest  = gid >> 5;
    int ch    = rest % nchunk;
    int mtile = rest / nchunk;
    int half_ = lane >> 4, idx16 = lane & 15;
    int m = mtile * 16 + idx16;

    v16bf vh, vl;
#pragma unroll
    for (int i = 0; i < 16; ++i) {
        int v = i >> 1, e = i & 1;
        int lk = ((v < 4) ? (2 * v) : (16 + 2 * (v - 4))) + 8 * half_ + e;
        int k  = ch * 32 + lk;
        float wv = 0.f;
        if (k < Ktot && m < Cout) {
            int c = k % CIN, t = k / CIN;      // constant divisors
            wv = w[((long)m * CIN + c) * KK + t];
        }
        float wh = bf_hi_part(wv);
        vh[i] = (__bf16)wh;
        vl[i] = (__bf16)(wv - wh);
    }
    v16bf* pv = (v16bf*)packed;
    pv[(size_t)gid * 2]     = vh;
    pv[(size_t)gid * 2 + 1] = vl;
}

// ---------------------------------------------------------------------------
// Fused (deformable) convolution as GEMM on the WMMA pipe, shape-specialized.
//   A = pre-packed bf16 hi/lo weights in fragment order (weights-stationary)
//   B = (bilinear-)gathered im2col, split to bf16 hi/lo on the fly
//   D = A*B + bias, optional gate multiply, optional ReLU.
// Grid: x = N-tile wave groups, y = M-groups (no runtime wid decomposition).
// Each wave holds MT accumulators (MT consecutive M-tiles) and reuses one
// gathered B fragment for all of them: 3*MT wmma per 32-wide K chunk.
// DEFORM=false -> plain conv, integer sampling, single load per element.
// ---------------------------------------------------------------------------
template<int CIN, int KSZ, bool DEFORM, int MT>
__global__ void __launch_bounds__(128)
conv_wmma_kernel(const float* __restrict__ x, const float* __restrict__ off,
                 const __bf16* __restrict__ wpack, const float* __restrict__ bias,
                 const float* __restrict__ gate, float* __restrict__ out,
                 int B, int H, int W, int Cout, int Mtiles, int stride, int pad,
                 int Ho, int Wo, int relu, int Ntiles)
{
    constexpr int KK     = KSZ * KSZ;
    constexpr int Ktot   = CIN * KK;
    constexpr int nchunk = (Ktot + 31) / 32;

    const int mgrp  = blockIdx.y;
    const int ntile = blockIdx.x * (blockDim.x >> 5) + (threadIdx.x >> 5);
    if (ntile >= Ntiles) return;             // wave-uniform: EXEC stays all-ones
    int lane  = threadIdx.x & 31;
    int half_ = lane >> 4;
    int idx16 = lane & 15;

    const int HW   = H * W;
    const int HoWo = Ho * Wo;
    const int Ntot = B * HoWo;

    int n_col = ntile * 16 + idx16;          // B column this lane gathers
    int nc = n_col < Ntot ? n_col : Ntot - 1;
    int b_ = nc / HoWo;
    int sp = nc % HoWo;
    int oy = sp / Wo;
    int ox = sp % Wo;
    const bool ncol_valid = (n_col < Ntot);
    const long xbase = (long)b_ * CIN * HW;

    // per-lane K offsets inside a 32-chunk (16-bit A/B fragment layout, ISA 7.12.2)
    int lk[16];
#pragma unroll
    for (int i = 0; i < 16; ++i) {
        int v = i >> 1, e = i & 1;
        int base = (v < 4) ? (2 * v) : (16 + 2 * (v - 4));
        lk[i] = base + 8 * half_ + e;
    }

    v8f zero = {};
    v8f acc[MT];
#pragma unroll
    for (int j = 0; j < MT; ++j) acc[j] = zero;

    const v16bf* wv = (const v16bf*)wpack;

    if constexpr (CIN % 32 == 0) {
        // One chunk == one kernel tap: hoist all tap geometry out of the i-loop.
        for (int ch = 0; ch < nchunk; ++ch) {
            const int kbase = ch * 32;
            const int t  = kbase / CIN;      // constant divisor -> shift
            const int cb = kbase % CIN;      // -> mask
            const int kh = t / KSZ, kw = t % KSZ;

            float c00 = 0.f, c01 = 0.f, c10 = 0.f, c11 = 0.f;
            int   i00 = 0, i01 = 0, i10 = 0, i11 = 0;   // DEFORM corners
            int   ii = 0; bool iv = false;              // plain-conv sample
            if constexpr (DEFORM) {
                long ob = (((long)b_ * (2 * KK) + 2 * t) * Ho + oy) * Wo + ox;
                float dy = off[ob], dx = off[ob + HoWo];
                float py = (float)(oy * stride - pad + kh) + dy;
                float px = (float)(ox * stride - pad + kw) + dx;
                float fy0 = floorf(py), fx0 = floorf(px);
                int y0 = (int)fy0, x0 = (int)fx0;
                float wy = py - fy0, wx = px - fx0;
                bool vy0 = (y0 >= 0) && (y0 < H);
                bool vy1 = (y0 + 1 >= 0) && (y0 + 1 < H);
                bool vx0 = (x0 >= 0) && (x0 < W);
                bool vx1 = (x0 + 1 >= 0) && (x0 + 1 < W);
                int yc0 = min(max(y0, 0), H - 1), yc1 = min(max(y0 + 1, 0), H - 1);
                int xc0 = min(max(x0, 0), W - 1), xc1 = min(max(x0 + 1, 0), W - 1);
                c00 = (vy0 && vx0) ? (1.f - wy) * (1.f - wx) : 0.f;
                c01 = (vy0 && vx1) ? (1.f - wy) * wx         : 0.f;
                c10 = (vy1 && vx0) ? wy * (1.f - wx)         : 0.f;
                c11 = (vy1 && vx1) ? wy * wx                 : 0.f;
                i00 = yc0 * W + xc0; i01 = yc0 * W + xc1;
                i10 = yc1 * W + xc0; i11 = yc1 * W + xc1;
            } else {
                int iy = oy * stride - pad + kh;
                int ix = ox * stride - pad + kw;
                iv = (iy >= 0) && (iy < H) && (ix >= 0) && (ix < W);
                ii = min(max(iy, 0), H - 1) * W + min(max(ix, 0), W - 1);
            }

            v16bf b_hi, b_lo;
#pragma unroll
            for (int i = 0; i < 16; ++i) {
                int c = cb + lk[i];
                float sv = 0.f;
                if (ncol_valid) {
                    const float* xc = x + xbase + (long)c * HW;
                    if constexpr (DEFORM)
                        sv = c00 * xc[i00] + c01 * xc[i01] + c10 * xc[i10] + c11 * xc[i11];
                    else
                        sv = iv ? xc[ii] : 0.f;
                }
                float sh = bf_hi_part(sv);
                b_hi[i] = (__bf16)sh; b_lo[i] = (__bf16)(sv - sh);
            }
#pragma unroll
            for (int j = 0; j < MT; ++j) {
                int mt = mgrp * MT + j;
                if (mt < Mtiles) {            // wave-uniform guard
                    size_t aoff = (((size_t)mt * nchunk + ch) * 32 + lane) * 2;
                    v16bf a_hi = wv[aoff], a_lo = wv[aoff + 1];
                    // bf16x3 fp32 emulation: hi*hi + lo*hi + hi*lo
                    acc[j] = __builtin_amdgcn_wmma_f32_16x16x32_bf16(false, a_hi, false, b_hi, (short)0, acc[j], false, false);
                    acc[j] = __builtin_amdgcn_wmma_f32_16x16x32_bf16(false, a_lo, false, b_hi, (short)0, acc[j], false, false);
                    acc[j] = __builtin_amdgcn_wmma_f32_16x16x32_bf16(false, a_hi, false, b_lo, (short)0, acc[j], false, false);
                }
            }
        }
    } else {
        // Generic small-CIN path (c1: CIN=3, Ktot=27) — constant divisors only.
        int   cur_t = -1;
        float c00 = 0.f, c01 = 0.f, c10 = 0.f, c11 = 0.f;
        int   i00 = 0, i01 = 0, i10 = 0, i11 = 0;
        int   ii = 0; bool iv = false;
        for (int ch = 0; ch < nchunk; ++ch) {
            const int kbase = ch * 32;
            v16bf b_hi, b_lo;
#pragma unroll
            for (int i = 0; i < 16; ++i) {
                int k = kbase + lk[i];
                float sv = 0.f;
                if (k < Ktot && ncol_valid) {
                    int c = k % CIN;     // constant divisor
                    int t = k / CIN;
                    if (t != cur_t) {
                        cur_t = t;
                        int kh = t / KSZ, kw = t % KSZ;
                        if constexpr (DEFORM) {
                            long ob = (((long)b_ * (2 * KK) + 2 * t) * Ho + oy) * Wo + ox;
                            float dy = off[ob], dx = off[ob + HoWo];
                            float py = (float)(oy * stride - pad + kh) + dy;
                            float px = (float)(ox * stride - pad + kw) + dx;
                            float fy0 = floorf(py), fx0 = floorf(px);
                            int y0 = (int)fy0, x0 = (int)fx0;
                            float wy = py - fy0, wx = px - fx0;
                            bool vy0 = (y0 >= 0) && (y0 < H);
                            bool vy1 = (y0 + 1 >= 0) && (y0 + 1 < H);
                            bool vx0 = (x0 >= 0) && (x0 < W);
                            bool vx1 = (x0 + 1 >= 0) && (x0 + 1 < W);
                            int yc0 = min(max(y0, 0), H - 1), yc1 = min(max(y0 + 1, 0), H - 1);
                            int xc0 = min(max(x0, 0), W - 1), xc1 = min(max(x0 + 1, 0), W - 1);
                            c00 = (vy0 && vx0) ? (1.f - wy) * (1.f - wx) : 0.f;
                            c01 = (vy0 && vx1) ? (1.f - wy) * wx         : 0.f;
                            c10 = (vy1 && vx0) ? wy * (1.f - wx)         : 0.f;
                            c11 = (vy1 && vx1) ? wy * wx                 : 0.f;
                            i00 = yc0 * W + xc0; i01 = yc0 * W + xc1;
                            i10 = yc1 * W + xc0; i11 = yc1 * W + xc1;
                        } else {
                            int iy = oy * stride - pad + kh;
                            int ix = ox * stride - pad + kw;
                            iv = (iy >= 0) && (iy < H) && (ix >= 0) && (ix < W);
                            ii = min(max(iy, 0), H - 1) * W + min(max(ix, 0), W - 1);
                        }
                    }
                    const float* xc = x + xbase + (long)c * HW;
                    if constexpr (DEFORM)
                        sv = c00 * xc[i00] + c01 * xc[i01] + c10 * xc[i10] + c11 * xc[i11];
                    else
                        sv = iv ? xc[ii] : 0.f;
                }
                float sh = bf_hi_part(sv);
                b_hi[i] = (__bf16)sh; b_lo[i] = (__bf16)(sv - sh);
            }
#pragma unroll
            for (int j = 0; j < MT; ++j) {
                int mt = mgrp * MT + j;
                if (mt < Mtiles) {
                    size_t aoff = (((size_t)mt * nchunk + ch) * 32 + lane) * 2;
                    v16bf a_hi = wv[aoff], a_lo = wv[aoff + 1];
                    acc[j] = __builtin_amdgcn_wmma_f32_16x16x32_bf16(false, a_hi, false, b_hi, (short)0, acc[j], false, false);
                    acc[j] = __builtin_amdgcn_wmma_f32_16x16x32_bf16(false, a_lo, false, b_hi, (short)0, acc[j], false, false);
                    acc[j] = __builtin_amdgcn_wmma_f32_16x16x32_bf16(false, a_hi, false, b_lo, (short)0, acc[j], false, false);
                }
            }
        }
    }

    // D layout: lanes 0-15 -> M = r, lanes 16-31 -> M = r + 8; N = lane % 16
#pragma unroll
    for (int j = 0; j < MT; ++j) {
        int mt = mgrp * MT + j;
#pragma unroll
        for (int r = 0; r < 8; ++r) {
            int cout = mt * 16 + r + 8 * half_;
            if (cout < Cout && ncol_valid) {
                float v = acc[j][r];
                long oidx = (((long)b_ * Cout + cout) * Ho + oy) * Wo + ox;
                if (bias) v += bias[cout];
                if (gate) v *= gate[oidx];
                if (relu) v = fmaxf(v, 0.f);
                out[oidx] = v;
            }
        }
    }
}

// Depthwise conv (groups == C): degenerate GEMM, bandwidth-bound -> VALU kernel.
template<int K, int DIL>
__global__ void __launch_bounds__(256)
dwconv_kernel(const float* __restrict__ x, const float* __restrict__ w,
              const float* __restrict__ bias, float* __restrict__ out,
              int B, int C, int H, int W, int pad)
{
    long n = (long)blockIdx.x * blockDim.x + threadIdx.x;
    long total = (long)B * C * H * W;
    if (n >= total) return;
    int ox = (int)(n % W); long t1 = n / W;
    int oy = (int)(t1 % H); long t2 = t1 / H;
    int c  = (int)(t2 % C); int b = (int)(t2 / C);
    const float* xc = x + ((long)b * C + c) * (long)H * W;
    const float* wc = w + (long)c * K * K;
    float acc = bias ? bias[c] : 0.f;
#pragma unroll
    for (int kh = 0; kh < K; ++kh) {
        int iy = oy - pad + kh * DIL;
        bool vy = (iy >= 0) && (iy < H);
        int iyc = min(max(iy, 0), H - 1);
#pragma unroll
        for (int kw = 0; kw < K; ++kw) {
            int ix = ox - pad + kw * DIL;
            bool v = vy && (ix >= 0) && (ix < W);
            int ixc = min(max(ix, 0), W - 1);
            acc += v ? wc[kh * K + kw] * xc[iyc * W + ixc] : 0.f;
        }
    }
    out[n] = acc;
}

template<int CIN, int KSZ, bool DEFORM, int MT>
static void run_conv(hipStream_t s, const float* x, const float* off,
                     const float* w, __bf16* wpack, const float* bias,
                     const float* gate, float* out, int B, int H, int W,
                     int Cout, int stride, int pad, int Ho, int Wo, int relu)
{
    constexpr int KK     = KSZ * KSZ;
    constexpr int Ktot   = CIN * KK;
    constexpr int nchunk = (Ktot + 31) / 32;
    int Mtiles = (Cout + 15) / 16;

    int ptotal = Mtiles * nchunk * 32;
    pack_w_kernel<CIN, KSZ><<<(ptotal + 255) / 256, 256, 0, s>>>(w, wpack, Cout, Mtiles);

    int Ntot    = B * Ho * Wo;
    int Ntiles  = (Ntot + 15) / 16;
    int Mgroups = (Mtiles + MT - 1) / MT;
    dim3 grid((Ntiles + 3) / 4, Mgroups);
    conv_wmma_kernel<CIN, KSZ, DEFORM, MT><<<grid, 128, 0, s>>>(
        x, off, wpack, bias, gate, out, B, H, W, Cout, Mtiles, stride, pad,
        Ho, Wo, relu, Ntiles);
}

template<int K, int DIL>
static void run_dw(hipStream_t s, const float* x, const float* w,
                   const float* bias, float* out, int B, int C, int H, int W, int pad)
{
    long total = (long)B * C * H * W;
    int blocks = (int)((total + 255) / 256);
    dwconv_kernel<K, DIL><<<blocks, 256, 0, s>>>(x, w, bias, out, B, C, H, W, pad);
}

extern "C" void kernel_launch(void* const* d_in, const int* in_sizes, int n_in,
                              void* d_out, int out_size, void* d_ws, size_t ws_size,
                              hipStream_t stream) {
    (void)in_sizes; (void)n_in; (void)out_size; (void)ws_size;
    const float* x0 = (const float*)d_in[0];
    // param order: c1(1..4) c2(5..8) lka1(9..14) c3(15..18) c4(19..22) lka2(23..28) c5(29..32) c6(33..36)
    const float *c1ow=(const float*)d_in[1],  *c1ob=(const float*)d_in[2],  *c1w=(const float*)d_in[3],  *c1b=(const float*)d_in[4];
    const float *c2ow=(const float*)d_in[5],  *c2ob=(const float*)d_in[6],  *c2w=(const float*)d_in[7],  *c2b=(const float*)d_in[8];
    const float *l1d5w=(const float*)d_in[9], *l1d5b=(const float*)d_in[10],*l1d7w=(const float*)d_in[11],*l1d7b=(const float*)d_in[12],
                *l1pww=(const float*)d_in[13],*l1pwb=(const float*)d_in[14];
    const float *c3ow=(const float*)d_in[15], *c3ob=(const float*)d_in[16], *c3w=(const float*)d_in[17], *c3b=(const float*)d_in[18];
    const float *c4ow=(const float*)d_in[19], *c4ob=(const float*)d_in[20], *c4w=(const float*)d_in[21], *c4b=(const float*)d_in[22];
    const float *l2d5w=(const float*)d_in[23],*l2d5b=(const float*)d_in[24],*l2d7w=(const float*)d_in[25],*l2d7b=(const float*)d_in[26],
                *l2pww=(const float*)d_in[27],*l2pwb=(const float*)d_in[28];
    const float *c5ow=(const float*)d_in[29], *c5ob=(const float*)d_in[30], *c5w=(const float*)d_in[31], *c5b=(const float*)d_in[32];
    const float *c6ow=(const float*)d_in[33], *c6ob=(const float*)d_in[34], *c6w=(const float*)d_in[35], *c6b=(const float*)d_in[36];
    float* outp = (float*)d_out;

    // workspace layout (floats)
    float* ws   = (float*)d_ws;
    float* bufA = ws;                    // 12,845,056  (max act: 4*64*224*224)
    float* bufB = bufA + 12845056;       // 12,845,056
    float* bufO = bufB + 12845056;       //  4,014,080  (max offsets: 4*18*224*224)
    float* bufT1= bufO + 4014080;        //  6,422,528  (4*128*112*112)
    float* bufT2= bufT1 + 6422528;       //  6,422,528
    __bf16* bufW= (__bf16*)(bufT2 + 6422528);  // 2 MB packed-weight slab (max 1.64 MB)

    const int Bn = 4;

    // c1: 3->64, K3 s1 p1 @224
    run_conv<3,3,false,2>(stream, x0, nullptr, c1ow, bufW, c1ob, nullptr, bufO, Bn,224,224, 18,1,1, 224,224, 0);
    run_conv<3,3,true ,4>(stream, x0, bufO,    c1w,  bufW, c1b,  nullptr, bufA, Bn,224,224, 64,1,1, 224,224, 1);
    // c2: 64->128, K5 s2 p2 @224 -> 112
    run_conv<64,5,false,4>(stream, bufA, nullptr, c2ow, bufW, c2ob, nullptr, bufO, Bn,224,224, 50,2,2, 112,112, 0);
    run_conv<64,5,true ,4>(stream, bufA, bufO,    c2w,  bufW, c2b,  nullptr, bufB, Bn,224,224, 128,2,2, 112,112, 1);
    // lka1 on bufB @112
    run_dw<5,1>(stream, bufB,  l1d5w, l1d5b, bufT1, Bn,128,112,112, 2);
    run_dw<7,3>(stream, bufT1, l1d7w, l1d7b, bufT2, Bn,128,112,112, 9);
    run_conv<128,1,false,4>(stream, bufT2, nullptr, l1pww, bufW, l1pwb, /*gate=*/bufB, bufA, Bn,112,112, 128,1,0, 112,112, 0);
    // c3: 128->128, K3 s1 p1 @112
    run_conv<128,3,false,2>(stream, bufA, nullptr, c3ow, bufW, c3ob, nullptr, bufO, Bn,112,112, 18,1,1, 112,112, 0);
    run_conv<128,3,true ,4>(stream, bufA, bufO,    c3w,  bufW, c3b,  nullptr, bufB, Bn,112,112, 128,1,1, 112,112, 1);
    // c4: 128->128, K5 s2 p2 @112 -> 56
    run_conv<128,5,false,4>(stream, bufB, nullptr, c4ow, bufW, c4ob, nullptr, bufO, Bn,112,112, 50,2,2, 56,56, 0);
    run_conv<128,5,true ,4>(stream, bufB, bufO,    c4w,  bufW, c4b,  nullptr, bufA, Bn,112,112, 128,2,2, 56,56, 1);
    // lka2 on bufA @56
    run_dw<5,1>(stream, bufA,  l2d5w, l2d5b, bufT1, Bn,128,56,56, 2);
    run_dw<7,3>(stream, bufT1, l2d7w, l2d7b, bufT2, Bn,128,56,56, 9);
    run_conv<128,1,false,4>(stream, bufT2, nullptr, l2pww, bufW, l2pwb, /*gate=*/bufA, bufB, Bn,56,56, 128,1,0, 56,56, 0);
    // c5: 128->64, K3 s1 p1 @56
    run_conv<128,3,false,2>(stream, bufB, nullptr, c5ow, bufW, c5ob, nullptr, bufO, Bn,56,56, 18,1,1, 56,56, 0);
    run_conv<128,3,true ,4>(stream, bufB, bufO,    c5w,  bufW, c5b,  nullptr, bufA, Bn,56,56, 64,1,1, 56,56, 1);
    // c6: 64->1, K1 s1 p0 @56 -> final output
    run_conv<64,1,false,1>(stream, bufA, nullptr, c6ow, bufW, c6ob, nullptr, bufO, Bn,56,56, 2,1,0, 56,56, 0);
    run_conv<64,1,true ,1>(stream, bufA, bufO,    c6w,  bufW, c6b,  nullptr, outp, Bn,56,56, 1,1,0, 56,56, 0);
}
